// RotaryMultiHeadAttention_62388694941896
// MI455X (gfx1250) — matrix operand order repaired and checked by reading, not compiled, not run on hardware
//
#include <hip/hip_runtime.h>

// =====================================================================
// RotaryMultiHeadAttention on MI455X (gfx1250, wave32, WMMA)
//
// ~52 GFLOP vs ~38 MB mandatory HBM traffic -> matrix-core bound.
// Intermediates (Q/K/V/attn ~34 MB f16) stay in the 192 MB L2.
// All GEMM stages: v_wmma_f32_16x16x32_f16 (f16 in, fp32 accum).
// Weight strips and attention K/V chunks staged into LDS with
// GLOBAL_LOAD_ASYNC_TO_LDS_B128 (ASYNCcnt pipeline, double-buffered
// in attention). GEMM K-loops fully unrolled with one-step-ahead
// A-fragment prefetch (no register ping-pong copies).
// =====================================================================

#define BATCH 4
#define SEQ   2048
#define DM    512
#define NH    8
#define HD    64
#define MTOT  (BATCH * SEQ)

typedef _Float16 f16;
typedef __attribute__((ext_vector_type(16))) _Float16 v16h;
typedef __attribute__((ext_vector_type(8)))  _Float16 v8h;
typedef __attribute__((ext_vector_type(8)))  float    v8f;
typedef int v4i __attribute__((vector_size(4 * sizeof(int))));
typedef __attribute__((address_space(1))) v4i* gptr_v4i;
typedef __attribute__((address_space(3))) v4i* sptr_v4i;

#if defined(__has_builtin)
#if __has_builtin(__builtin_amdgcn_global_load_async_to_lds_b128)
#define HAVE_ASYNC_LDS 1
#endif
#endif
#ifndef HAVE_ASYNC_LDS
#define HAVE_ASYNC_LDS 0
#endif

static __device__ __forceinline__ v8f wmma16(v16h a, v16h b, v8f c) {
  return __builtin_amdgcn_wmma_f32_16x16x32_f16(false, a, false, b, (short)0, c,
                                                false, false);
}

// A-fragment (16x32 f16, MxK) from row-major src[m][ld] (global or LDS).
static __device__ __forceinline__ v16h a_frag(const f16* src, int ld, int m0, int k0) {
  const int lane = threadIdx.x & 31;
  const int m    = m0 + (lane & 15);
  const int ko   = k0 + ((lane >> 4) << 4);
  return *(const v16h*)(src + (size_t)m * ld + ko);
}

// B-fragment (32x16 f16, KxN) where B[k][n] = src[n][k] (src row-major).
static __device__ __forceinline__ v16h b_frag(const f16* src, int ld, int k0, int n0) {
  const int lane = threadIdx.x & 31;
  const int n    = n0 + (lane & 15);
  const int kb   = k0 + ((lane >> 4) << 4);
  return *(const v16h*)(src + (size_t)n * ld + kb);
}

// 16-byte global -> LDS copy: async (ASYNCcnt) when available.
static __device__ __forceinline__ void cp16_g2s(f16* ldst, const f16* gsrc) {
#if HAVE_ASYNC_LDS
  __builtin_amdgcn_global_load_async_to_lds_b128((gptr_v4i)gsrc, (sptr_v4i)ldst,
                                                 0, 0);
#else
  *(v8h*)ldst = *(const v8h*)gsrc;
#endif
}

static __device__ __forceinline__ void wait_async_le4() {
#if HAVE_ASYNC_LDS
#if __has_builtin(__builtin_amdgcn_s_wait_asynccnt)
  __builtin_amdgcn_s_wait_asynccnt(4);
#else
  asm volatile("s_wait_asynccnt 0x4" ::: "memory");
#endif
#endif
}
static __device__ __forceinline__ void wait_async_le0() {
#if HAVE_ASYNC_LDS
#if __has_builtin(__builtin_amdgcn_s_wait_asynccnt)
  __builtin_amdgcn_s_wait_asynccnt(0);
#else
  asm volatile("s_wait_asynccnt 0x0" ::: "memory");
#endif
#endif
}

// ---------------------------------------------------------------------
// fp32 -> f16 conversion, 8 elements/thread
// ---------------------------------------------------------------------
__global__ void cvt_kernel(const float* __restrict__ s, f16* __restrict__ d, int n) {
  int i = (blockIdx.x * blockDim.x + threadIdx.x) * 8;
  if (i < n) {
    float4 a = *(const float4*)(s + i);
    float4 b = *(const float4*)(s + i + 4);
    v8h o;
    o[0] = (f16)a.x; o[1] = (f16)a.y; o[2] = (f16)a.z; o[3] = (f16)a.w;
    o[4] = (f16)b.x; o[5] = (f16)b.y; o[6] = (f16)b.z; o[7] = (f16)b.w;
    *(v8h*)(d + i) = o;
  }
}

// ---------------------------------------------------------------------
// QKV projection + bias + RoPE.
// grid = (MTOT/128, NH, 3[q,k,v]); block = 256 (8 waves).
// 64-row weight strip (64 KB, contiguous) async-staged into LDS once;
// K-loop fully unrolled, A-fragment prefetched one step ahead.
// ---------------------------------------------------------------------
__global__ __launch_bounds__(256)
void qkv_rope_kernel(const f16* __restrict__ xb,
                     const f16* __restrict__ Wq16, const f16* __restrict__ Wk16,
                     const f16* __restrict__ Wv16,
                     const float* __restrict__ bq, const float* __restrict__ bk,
                     const float* __restrict__ bv,
                     const float* __restrict__ posq, const float* __restrict__ posk,
                     f16* __restrict__ Qh, f16* __restrict__ Kh, f16* __restrict__ Vt) {
  __shared__ alignas(32) f16 Wl[64 * DM];            // [n_local][k], 64 KB

  const int z    = blockIdx.z;
  const f16*   W    = (z == 0) ? Wq16 : (z == 1) ? Wk16 : Wv16;
  const float* bias = (z == 0) ? bq   : (z == 1) ? bk   : bv;
  const float* pos  = (z == 1) ? posk : posq;

  const int tid  = threadIdx.x;
  const int wave = tid >> 5;
  const int lane = tid & 31;
  const int half = lane >> 4;
  const int m0   = blockIdx.x * 128 + wave * 16;
  const int h    = blockIdx.y;
  const int n0   = h * HD;

  // stage weight strip: rows n0..n0+63 are contiguous 64 KB in global
  {
    const f16* gW = W + (size_t)n0 * DM;
#pragma unroll
    for (int c = 0; c < 16; ++c) {
      int idx = (c * 256 + tid) * 8;                 // f16 offset, 16B chunks
      cp16_g2s(&Wl[idx], gW + idx);
    }
  }

  const v8f vzero = {0.f,0.f,0.f,0.f,0.f,0.f,0.f,0.f};
  v8f acc[4] = {vzero, vzero, vzero, vzero};

  wait_async_le0();
  __syncthreads();

  // fully unrolled K-loop, software-pipelined A loads (SSA, no copies)
  v16h a_cur = a_frag(xb, DM, m0, 0);
#pragma unroll
  for (int kk = 0; kk < 16; ++kk) {
    v16h a_nxt = a_cur;
    if (kk < 15) a_nxt = a_frag(xb, DM, m0, (kk + 1) * 32);
#pragma unroll
    for (int t = 0; t < 4; ++t)
      acc[t] = wmma16(a_cur, b_frag(Wl, DM, kk * 32, t * 16), acc[t]);
    a_cur = a_nxt;
  }

#pragma unroll
  for (int t = 0; t < 4; ++t) {
    float bb = bias[n0 + t * 16 + (lane & 15)];
#pragma unroll
    for (int r = 0; r < 8; ++r) acc[t][r] += bb;
  }

  // RoPE: partner channel d^1 lives in lane^1 of the C layout
  if (z < 2) {
    float posr[8];
#pragma unroll
    for (int r = 0; r < 8; ++r) {
      int mtok = m0 + r + 8 * half;
      float2 pp = *(const float2*)(pos + (size_t)mtok * 2);
      posr[r] = pp.x + pp.y;
    }
#pragma unroll
    for (int t = 0; t < 4; ++t) {
      int d = t * 16 + (lane & 15);
      int p = d >> 1;
      float inv_freq = __expf(-(float)p * (9.210340371976184f / 16.0f));
      float sgn = (d & 1) ? 1.0f : -1.0f;
#pragma unroll
      for (int r = 0; r < 8; ++r) {
        float c = acc[t][r];
        float o = __shfl_xor(c, 1, 32);
        float sn, cs;
        __sincosf(posr[r] * inv_freq, &sn, &cs);
        acc[t][r] = c * cs + o * sgn * sn;
      }
    }
  }

  const int b    = m0 / SEQ;
  const int ntok = m0 % SEQ;
  if (z == 2) {
#pragma unroll
    for (int t = 0; t < 4; ++t) {
      int d = t * 16 + (lane & 15);
      v8h pk;
#pragma unroll
      for (int r = 0; r < 8; ++r) pk[r] = (f16)acc[t][r];
      f16* dst = Vt + ((((size_t)b * NH + h) * HD + d) * SEQ) + ntok + 8 * half;
      *(v8h*)dst = pk;
    }
  } else {
    f16* Dst = (z == 0) ? Qh : Kh;
#pragma unroll
    for (int t = 0; t < 4; ++t) {
      int d = t * 16 + (lane & 15);
#pragma unroll
      for (int r = 0; r < 8; ++r) {
        int tok = ntok + r + 8 * half;
        Dst[(((size_t)b * NH + h) * SEQ + tok) * HD + d] = (f16)acc[t][r];
      }
    }
  }
}

// ---------------------------------------------------------------------
// Flash attention. grid = (SEQ/128, NH, BATCH); block = 256 (8 waves).
// K/V chunks (64 tokens, 8 KB each) async-staged into LDS per BLOCK,
// explicitly double-buffered (2x-unrolled pipeline, static buffer ids).
// ---------------------------------------------------------------------
__global__ __launch_bounds__(256)
void attn_kernel(const f16* __restrict__ Qh, const f16* __restrict__ Kh,
                 const f16* __restrict__ Vt, f16* __restrict__ attn) {
  __shared__ alignas(32) f16 Kt[2][64 * HD];     // [token][d], 8 KB per buf
  __shared__ alignas(32) f16 Vts[2][HD * 64];    // [d][token], 8 KB per buf
  __shared__ alignas(32) f16 lds_p[8][16 * 64];  // P scratch, 2 KB per wave

  const int tid  = threadIdx.x;
  const int wave = tid >> 5;
  const int lane = tid & 31;
  const int half = lane >> 4;
  const int h = blockIdx.y, b = blockIdx.z;
  const int q0 = blockIdx.x * 128 + wave * 16;

  const f16* Qbh = Qh + ((size_t)b * NH + h) * SEQ * HD;
  const f16* Kbh = Kh + ((size_t)b * NH + h) * SEQ * HD;
  const f16* Vbh = Vt + ((size_t)b * NH + h) * HD * SEQ;
  f16* myP = &lds_p[wave][0];

  const int vrow = tid >> 3, vseg = tid & 7;     // V staging coordinates

  auto stage = [&](int buf, int kc) {
    const f16* gK = Kbh + (size_t)kc * HD;       // contiguous 8 KB
    f16* sK = &Kt[buf][0];
    cp16_g2s(sK + tid * 8,        gK + tid * 8);
    cp16_g2s(sK + 2048 + tid * 8, gK + 2048 + tid * 8);
    const f16* gV = Vbh + (size_t)vrow * SEQ + kc + vseg * 8;
    f16* sV = &Vts[buf][0] + vrow * 64 + vseg * 8;
    cp16_g2s(sV,           gV);
    cp16_g2s(sV + 32 * 64, gV + (size_t)32 * SEQ);
  };

  // Q fragments, pre-scaled by 1/sqrt(64)
  v16h aq0 = a_frag(Qbh, HD, q0, 0);
  v16h aq1 = a_frag(Qbh, HD, q0, 32);
#pragma unroll
  for (int i = 0; i < 16; ++i) {
    aq0[i] = (f16)((float)aq0[i] * 0.125f);
    aq1[i] = (f16)((float)aq1[i] * 0.125f);
  }

  const v8f vzero = {0.f,0.f,0.f,0.f,0.f,0.f,0.f,0.f};
  v8f o[4] = {vzero, vzero, vzero, vzero};
  float mrun[8], lrun[8];
#pragma unroll
  for (int r = 0; r < 8; ++r) { mrun[r] = -1e30f; lrun[r] = 0.f; }

  // one 64-token chunk: S = QK^T, online softmax, O += P V
  auto do_chunk = [&](const f16* KtC, const f16* VtC) {
    v8f s[4];
#pragma unroll
    for (int t = 0; t < 4; ++t) {
      v16h bk0 = b_frag(KtC, HD, 0,  t * 16);
      v16h bk1 = b_frag(KtC, HD, 32, t * 16);
      v8f acc = wmma16(aq0, bk0, vzero);
      s[t] = wmma16(aq1, bk1, acc);
    }
    float mnew[8], alpha[8];
#pragma unroll
    for (int r = 0; r < 8; ++r) {
      float cm = fmaxf(fmaxf(s[0][r], s[1][r]), fmaxf(s[2][r], s[3][r]));
      cm = fmaxf(cm, __shfl_xor(cm, 1, 32));
      cm = fmaxf(cm, __shfl_xor(cm, 2, 32));
      cm = fmaxf(cm, __shfl_xor(cm, 4, 32));
      cm = fmaxf(cm, __shfl_xor(cm, 8, 32));
      mnew[r]  = fmaxf(mrun[r], cm);
      alpha[r] = __expf(mrun[r] - mnew[r]);
      mrun[r]  = mnew[r];
      lrun[r] *= alpha[r];
    }
#pragma unroll
    for (int dt = 0; dt < 4; ++dt)
#pragma unroll
      for (int r = 0; r < 8; ++r) o[dt][r] *= alpha[r];

#pragma unroll
    for (int r = 0; r < 8; ++r) {
      float rs = 0.f;
      int row = r + 8 * half;
#pragma unroll
      for (int t = 0; t < 4; ++t) {
        float p = __expf(s[t][r] - mnew[r]);
        rs += p;
        myP[row * 64 + t * 16 + (lane & 15)] = (f16)p;
      }
      rs += __shfl_xor(rs, 1, 32);
      rs += __shfl_xor(rs, 2, 32);
      rs += __shfl_xor(rs, 4, 32);
      rs += __shfl_xor(rs, 8, 32);
      lrun[r] += rs;
    }

    v16h pa0 = a_frag(myP, 64, 0, 0);
    v16h pa1 = a_frag(myP, 64, 0, 32);
#pragma unroll
    for (int dt = 0; dt < 4; ++dt) {
      v16h bv0 = b_frag(VtC, 64, 0,  dt * 16);
      v16h bv1 = b_frag(VtC, 64, 32, dt * 16);
      o[dt] = wmma16(pa0, bv0, o[dt]);
      o[dt] = wmma16(pa1, bv1, o[dt]);
    }
  };

  stage(0, 0);
  for (int kc = 0; kc < SEQ; kc += 128) {          // 2 chunks per iter
    // phase A: buffer 0 holds chunk kc; prefetch kc+64 into buffer 1
    if (kc + 64 < SEQ) { stage(1, kc + 64); wait_async_le4(); }
    else               { wait_async_le0(); }
    __syncthreads();
    do_chunk(&Kt[0][0], &Vts[0][0]);
    __syncthreads();
    // phase B: buffer 1 holds chunk kc+64; prefetch kc+128 into buffer 0
    if (kc + 128 < SEQ) { stage(0, kc + 128); wait_async_le4(); }
    else                { wait_async_le0(); }
    __syncthreads();
    do_chunk(&Kt[1][0], &Vts[1][0]);
    __syncthreads();
  }

  // ---- epilogue ----
#pragma unroll
  for (int dt = 0; dt < 4; ++dt) {
    int d = dt * 16 + (lane & 15);
#pragma unroll
    for (int r = 0; r < 8; ++r) {
      int tok = q0 + r + 8 * half;
      float val = o[dt][r] / lrun[r];
      attn[((size_t)b * SEQ + tok) * DM + h * HD + d] = (f16)val;
    }
  }
}

// ---------------------------------------------------------------------
// Output projection: out = attn @ Wo^T + bo (fp32).
// Same LDS weight staging + fully unrolled, prefetched K-loop.
// ---------------------------------------------------------------------
__global__ __launch_bounds__(256)
void out_proj_kernel(const f16* __restrict__ attn, const f16* __restrict__ Wo16,
                     const float* __restrict__ bo, float* __restrict__ out) {
  __shared__ alignas(32) f16 Wl[64 * DM];          // 64 KB

  const int tid  = threadIdx.x;
  const int wave = tid >> 5;
  const int lane = tid & 31;
  const int half = lane >> 4;
  const int m0 = blockIdx.x * 128 + wave * 16;
  const int n0 = blockIdx.y * 64;

  {
    const f16* gW = Wo16 + (size_t)n0 * DM;
#pragma unroll
    for (int c = 0; c < 16; ++c) {
      int idx = (c * 256 + tid) * 8;
      cp16_g2s(&Wl[idx], gW + idx);
    }
  }

  const v8f vzero = {0.f,0.f,0.f,0.f,0.f,0.f,0.f,0.f};
  v8f acc[4] = {vzero, vzero, vzero, vzero};

  wait_async_le0();
  __syncthreads();

  v16h a_cur = a_frag(attn, DM, m0, 0);
#pragma unroll
  for (int kk = 0; kk < 16; ++kk) {
    v16h a_nxt = a_cur;
    if (kk < 15) a_nxt = a_frag(attn, DM, m0, (kk + 1) * 32);
#pragma unroll
    for (int t = 0; t < 4; ++t)
      acc[t] = wmma16(a_cur, b_frag(Wl, DM, kk * 32, t * 16), acc[t]);
    a_cur = a_nxt;
  }

#pragma unroll
  for (int t = 0; t < 4; ++t) {
    int n = n0 + t * 16 + (lane & 15);
    float bb = bo[n];
#pragma unroll
    for (int r = 0; r < 8; ++r)
      out[(size_t)(m0 + r + 8 * half) * DM + n] = acc[t][r] + bb;
  }
}

// ---------------------------------------------------------------------
extern "C" void kernel_launch(void* const* d_in, const int* in_sizes, int n_in,
                              void* d_out, int out_size, void* d_ws, size_t ws_size,
                              hipStream_t stream) {
  const float* x   = (const float*)d_in[0];
  const float* qp  = (const float*)d_in[1];
  const float* kp  = (const float*)d_in[2];
  const float* Wq  = (const float*)d_in[3];
  const float* bq  = (const float*)d_in[4];
  const float* Wk  = (const float*)d_in[5];
  const float* bk  = (const float*)d_in[6];
  const float* Wv  = (const float*)d_in[7];
  const float* bv  = (const float*)d_in[8];
  const float* Wo  = (const float*)d_in[9];
  const float* bo  = (const float*)d_in[10];
  float* out = (float*)d_out;

  char* ws = (char*)d_ws;
  f16* xb   = (f16*)(ws + 0);                                  // 8 MiB
  f16* Wq16 = (f16*)(ws + (size_t)8 * 1024 * 1024);
  f16* Wk16 = (f16*)(ws + (size_t)8 * 1024 * 1024 + 524288);
  f16* Wv16 = (f16*)(ws + (size_t)8 * 1024 * 1024 + 1048576);
  f16* Wo16 = (f16*)(ws + (size_t)8 * 1024 * 1024 + 1572864);
  f16* Qh   = (f16*)(ws + (size_t)10 * 1024 * 1024);           // 8 MiB
  f16* Kh   = (f16*)(ws + (size_t)18 * 1024 * 1024);           // 8 MiB
  f16* Vt   = (f16*)(ws + (size_t)26 * 1024 * 1024);           // 8 MiB
  f16* attn = (f16*)(ws + (size_t)34 * 1024 * 1024);           // 8 MiB

  {
    int n = MTOT * DM;
    cvt_kernel<<<dim3((n / 8 + 255) / 256), dim3(256), 0, stream>>>(x, xb, n);
    n = DM * DM;
    dim3 g((n / 8 + 255) / 256);
    cvt_kernel<<<g, dim3(256), 0, stream>>>(Wq, Wq16, n);
    cvt_kernel<<<g, dim3(256), 0, stream>>>(Wk, Wk16, n);
    cvt_kernel<<<g, dim3(256), 0, stream>>>(Wv, Wv16, n);
    cvt_kernel<<<g, dim3(256), 0, stream>>>(Wo, Wo16, n);
  }

  qkv_rope_kernel<<<dim3(MTOT / 128, NH, 3), dim3(256), 0, stream>>>(
      xb, Wq16, Wk16, Wv16, bq, bk, bv, qp, kp, Qh, Kh, Vt);

  attn_kernel<<<dim3(SEQ / 128, NH, BATCH), dim3(256), 0, stream>>>(Qh, Kh, Vt, attn);

  out_proj_kernel<<<dim3(MTOT / 128, DM / 64), dim3(256), 0, stream>>>(attn, Wo16, bo, out);
}
